// LinearAttention_79267916415318
// MI455X (gfx1250) — matrix operand order repaired and checked by reading, not compiled, not run on hardware
//
#include <hip/hip_runtime.h>
#include <hip/hip_bf16.h>

// ---------------------------------------------------------------------------
// Linear attention for MI455X (gfx1250, wave32, WMMA bf16 -> f32 accumulate)
// Async global->LDS staging (ASYNCcnt) + transposed LDS layouts so every
// WMMA fragment read vectorizes to ds_load_b128.
// ---------------------------------------------------------------------------

typedef __bf16 bf16_t;
typedef __attribute__((ext_vector_type(16))) __bf16 v16bf;
typedef __attribute__((ext_vector_type(8)))  float  v8f;

#define NUM_HEADS 8
#define HEAD_DIM  64
#define EPS       1e-4f

#define B_     16
#define N_     4096
#define DIM_   512
#define INNER_ 512              // NUM_HEADS * HEAD_DIM
#define ROWS_  (B_ * N_)        // 65536

// ------------------------- workspace layout (bytes) ------------------------
#define SZ_HEADS   ((size_t)B_ * NUM_HEADS * N_ * HEAD_DIM * 2)   // 67108864
#define OFF_QH     ((size_t)0)
#define OFF_KH     (OFF_QH + SZ_HEADS)
#define OFF_VH     (OFF_KH + SZ_HEADS)
#define OFF_ATTN   (OFF_VH + SZ_HEADS)                            // bf16 [B][N][INNER]
#define OFF_WQKV   (OFF_ATTN + (size_t)B_ * N_ * INNER_ * 2)
#define OFF_WOUT   (OFF_WQKV + (size_t)3 * INNER_ * DIM_ * 2)
#define OFF_KV     (OFF_WOUT + (size_t)DIM_ * INNER_ * 2)         // f32 [128][64][64]
#define OFF_KSUM   (OFF_KV + (size_t)B_ * NUM_HEADS * HEAD_DIM * HEAD_DIM * 4)

// ------------------------- async global -> LDS copy ------------------------
// CDNA5 GLOBAL_LOAD_ASYNC_TO_LDS_B128 (tracked with ASYNCcnt). LDS byte
// address = low 32 bits of the generic shared pointer on AMDGPU.
__device__ __forceinline__ void async_copy_b128(const void* gaddr, void* lds) {
  unsigned lds_off = (unsigned)(uintptr_t)lds;
  asm volatile("global_load_async_to_lds_b128 %0, %1, off"
               :: "v"(lds_off), "v"(gaddr) : "memory");
}
__device__ __forceinline__ void async_wait0() {
  asm volatile("s_wait_asynccnt 0x0" ::: "memory");
}

// ----------------------------- WMMA helpers --------------------------------
__device__ __forceinline__ v8f wmma_bf16(v16bf a, v16bf b, v8f c) {
  return __builtin_amdgcn_wmma_f32_16x16x32_bf16(false, a, false, b, (short)0, c,
                                                 false, false);
}

// A-frag (16x32 bf16, MxK): lane L: row m = L&15, K-half h = L>>4.
// A[m][k] = src[m*mstride + k*kstride].
__device__ __forceinline__ v16bf load_a_frag(const bf16_t* src, int mstride,
                                             int kstride) {
  const int lane = threadIdx.x & 31;
  const int m = lane & 15, h = lane >> 4;
  v16bf a;
#pragma unroll
  for (int i = 0; i < 8; ++i) {
    const int k = ((i < 4) ? 0 : 16) + h * 8 + 2 * (i & 3);
    a[2 * i]     = src[m * mstride + k * kstride];
    a[2 * i + 1] = src[m * mstride + (k + 1) * kstride];
  }
  return a;
}

// B-frag (32x16 bf16, KxN): lane L: column n = L&15, K-half h = L>>4 (same k order as A).
// B[k][n] = src[k*kstride + n*nstride].
__device__ __forceinline__ v16bf load_b_frag(const bf16_t* src, int kstride,
                                             int nstride) {
  const int lane = threadIdx.x & 31;
  const int n = lane & 15, h = lane >> 4;
  v16bf b;
#pragma unroll
  for (int i = 0; i < 8; ++i) {
    const int k = ((i < 4) ? 0 : 16) + h * 8 + 2 * (i & 3);
    b[2 * i]     = src[k * kstride + n * nstride];
    b[2 * i + 1] = src[(k + 1) * kstride + n * nstride];
  }
  return b;
}

// ------------------------- weight conversion -------------------------------
__global__ __launch_bounds__(256) void cvt_f32_bf16_kernel(
    const float* __restrict__ in, bf16_t* __restrict__ out, int n) {
  int i = blockIdx.x * blockDim.x + threadIdx.x;
  if (i < n) out[i] = (bf16_t)in[i];
}

// ---------------------------------------------------------------------------
// K1: qkv = x @ W_qkv^T ; activation on q,k ; scatter into [b,h,n,d] layouts.
// Block tile 128x128, K-step 32; 8 waves, each 32x64 (2x4 WMMA tiles).
// ---------------------------------------------------------------------------
__global__ __launch_bounds__(256) void qkv_gemm_kernel(
    const float* __restrict__ x, const bf16_t* __restrict__ wq,
    bf16_t* __restrict__ qh, bf16_t* __restrict__ kh, bf16_t* __restrict__ vh) {
  __shared__ bf16_t As[128 * 32];
  __shared__ bf16_t Bs[128 * 32];

  const int tid = threadIdx.x;
  const int m0 = blockIdx.x * 128;
  const int j0 = blockIdx.y * 128;
  const int wid = tid >> 5;
  const int wm = wid & 3;        // 4 wave rows of 32
  const int wn = wid >> 2;       // 2 wave cols of 64

  v8f acc[2][4] = {};

  for (int k0 = 0; k0 < DIM_; k0 += 32) {
    // stage W tile (bf16, row-major): async copy 128x32 = 512 x b128
#pragma unroll
    for (int t = 0; t < 2; ++t) {
      const int idx = tid + t * 256;
      const int row = idx >> 2, c = (idx & 3) * 8;
      async_copy_b128(wq + (size_t)(j0 + row) * DIM_ + k0 + c,
                      &Bs[row * 32 + c]);
    }
    // stage x tile (f32 -> bf16): 128x32 floats = 1024 float4
#pragma unroll
    for (int t = 0; t < 4; ++t) {
      const int idx = tid + t * 256;
      const int row = idx >> 3, c4 = (idx & 7) * 4;
      const float* gp = x + (size_t)(m0 + row) * DIM_ + k0 + c4;
      if (k0 + 32 < DIM_) __builtin_prefetch(gp + 32, 0, 1);
      const float4 f = *(const float4*)gp;
      bf16_t* d = &As[row * 32 + c4];
      d[0] = (bf16_t)f.x; d[1] = (bf16_t)f.y;
      d[2] = (bf16_t)f.z; d[3] = (bf16_t)f.w;
    }
    async_wait0();
    __syncthreads();

    v16bf a0 = load_a_frag(&As[(wm * 32) * 32], 32, 1);
    v16bf a1 = load_a_frag(&As[(wm * 32 + 16) * 32], 32, 1);
#pragma unroll
    for (int j = 0; j < 4; ++j) {
      v16bf b = load_b_frag(&Bs[(wn * 64 + j * 16) * 32], 1, 32);
      acc[0][j] = wmma_bf16(a0, b, acc[0][j]);
      acc[1][j] = wmma_bf16(a1, b, acc[1][j]);
    }
    __syncthreads();
  }

  // epilogue: 1+elu on q,k ; scatter to [b,h,n,d]
  const int lane = tid & 31;
  const int nloc = lane & 15, half = lane >> 4;
#pragma unroll
  for (int i = 0; i < 2; ++i)
#pragma unroll
    for (int j = 0; j < 4; ++j)
#pragma unroll
      for (int r = 0; r < 8; ++r) {
        const int mloc = wm * 32 + i * 16 + r + 8 * half;
        const int jloc = wn * 64 + j * 16 + nloc;
        const int M = m0 + mloc;
        const int J = j0 + jloc;
        const int bi = M >> 12, nn = M & (N_ - 1);
        const int sec = J >> 9;            // 0=q, 1=k, 2=v
        const int col = J & (INNER_ - 1);
        const int h = col >> 6, d = col & (HEAD_DIM - 1);
        float v = acc[i][j][r];
        if (sec < 2) v = (v > 0.f) ? (v + 1.f) : __expf(v);
        bf16_t* dst = (sec == 0) ? qh : (sec == 1) ? kh : vh;
        dst[(size_t)((bi * NUM_HEADS + h) * N_ + nn) * HEAD_DIM + d] = (bf16_t)v;
      }
}

// ---------------------------------------------------------------------------
// K2: per (b,h): kv[d][e] = sum_n k[n][d]*v[n][e] ; ksum[d] = sum_n k[n][d]
// k/v staged TRANSPOSED in LDS (Kt[d][n], Vt[e][n]) so fragments are
// contiguous ds_load_b128 reads.
// ---------------------------------------------------------------------------
__global__ __launch_bounds__(256) void kv_kernel(
    const bf16_t* __restrict__ kh, const bf16_t* __restrict__ vh,
    float* __restrict__ kvbuf, float* __restrict__ ksumbuf) {
  __shared__ bf16_t Kt[64 * 32];   // [d][n]
  __shared__ bf16_t Vt[64 * 32];   // [e][n]

  const int tid = threadIdx.x;
  const int bh = blockIdx.x;
  const bf16_t* kbase = kh + (size_t)bh * N_ * HEAD_DIM;
  const bf16_t* vbase = vh + (size_t)bh * N_ * HEAD_DIM;
  const int wid = tid >> 5;
  const int t0 = wid * 2;

  v8f acc[2] = {};
  float ks = 0.f;

  for (int n0 = 0; n0 < N_; n0 += 32) {
    const int g = tid & 127;
    const bf16_t* src = (tid < 128) ? kbase : vbase;
    bf16_t* dstT = (tid < 128) ? Kt : Vt;
#pragma unroll
    for (int kk = 0; kk < 2; ++kk) {
      const int idx = g + kk * 128;       // 0..255
      const int row = idx >> 3;           // n-local 0..31
      const int c = (idx & 7) * 8;        // d/e base
      uint4 u = *(const uint4*)(src + (size_t)(n0 + row) * HEAD_DIM + c);
      const bf16_t* e8 = (const bf16_t*)&u;
#pragma unroll
      for (int q = 0; q < 8; ++q) dstT[(c + q) * 32 + row] = e8[q];
    }
    __syncthreads();

#pragma unroll
    for (int s = 0; s < 2; ++s) {
      const int t = t0 + s;
      const int ti = t >> 2, tj = t & 3;
      // A = k^T : A[m=d][k=n] = Kt[(ti*16+m)*32 + n]  (contiguous in k)
      v16bf a = load_a_frag(&Kt[(ti * 16) * 32], 32, 1);
      // B = v   : B[k=n][nn=e] = Vt[(tj*16+nn)*32 + n] (contiguous in k)
      v16bf b = load_b_frag(&Vt[(tj * 16) * 32], 1, 32);
      acc[s] = wmma_bf16(a, b, acc[s]);
    }
    if (tid < 64) {                       // ksum: contiguous row of Kt
#pragma unroll
      for (int r = 0; r < 32; ++r) ks += (float)Kt[tid * 32 + r];
    }
    __syncthreads();
  }

  const int lane = tid & 31, nloc = lane & 15, half = lane >> 4;
#pragma unroll
  for (int s = 0; s < 2; ++s) {
    const int t = t0 + s;
    const int ti = t >> 2, tj = t & 3;
#pragma unroll
    for (int r = 0; r < 8; ++r) {
      const int d = ti * 16 + r + 8 * half;
      const int e = tj * 16 + nloc;
      kvbuf[(size_t)bh * 4096 + d * 64 + e] = acc[s][r];
    }
  }
  if (tid < 64) ksumbuf[bh * 64 + tid] = ks;
}

// ---------------------------------------------------------------------------
// K3: out_head = (q @ kv) * z ; z = 1/max(q . ksum, eps)
// Q staged via async copy; kv staged TRANSPOSED (KVt[e][k]) for contiguous
// B-fragment reads.
// ---------------------------------------------------------------------------
__global__ __launch_bounds__(256) void attn_out_kernel(
    const bf16_t* __restrict__ qh, const float* __restrict__ kvbuf,
    const float* __restrict__ ksumbuf, bf16_t* __restrict__ attn) {
  __shared__ bf16_t Qs[128 * 64];
  __shared__ bf16_t KVt[64 * 64];   // [e][k]
  __shared__ float ksum_l[64];
  __shared__ float zl[128];

  const int tid = threadIdx.x;
  const int n0 = blockIdx.x * 128;
  const int bh = blockIdx.y;
  const bf16_t* qbase = qh + (size_t)bh * N_ * HEAD_DIM + (size_t)n0 * HEAD_DIM;
  const float* kvsrc = kvbuf + (size_t)bh * 4096;

  // stage q (bf16, row-major): async 128x64 = 1024 x b128
#pragma unroll
  for (int t = 0; t < 4; ++t) {
    const int idx = tid + t * 256;
    const int row = idx >> 3, c = (idx & 7) * 8;
    async_copy_b128(qbase + (size_t)row * HEAD_DIM + c, &Qs[row * 64 + c]);
  }
  // stage kv (f32 -> bf16, transposed): kv row-major [k][e]
#pragma unroll
  for (int t = 0; t < 4; ++t) {
    const int idx = tid + t * 256;      // float4 index
    const int kk = idx >> 4;            // k row 0..63
    const int e = (idx & 15) * 4;       // e base
    const float4 f = *(const float4*)(kvsrc + idx * 4);
    KVt[(e + 0) * 64 + kk] = (bf16_t)f.x;
    KVt[(e + 1) * 64 + kk] = (bf16_t)f.y;
    KVt[(e + 2) * 64 + kk] = (bf16_t)f.z;
    KVt[(e + 3) * 64 + kk] = (bf16_t)f.w;
  }
  if (tid < 64) ksum_l[tid] = ksumbuf[bh * 64 + tid];
  async_wait0();
  __syncthreads();

  if (tid < 128) {
    float dot = 0.f;
#pragma unroll
    for (int d = 0; d < HEAD_DIM; ++d) dot += (float)Qs[tid * 64 + d] * ksum_l[d];
    zl[tid] = 1.f / fmaxf(dot, EPS);
  }
  __syncthreads();

  const int wid = tid >> 5, wm = wid & 3, wn = wid >> 2;
  v8f acc[2][2] = {};
#pragma unroll
  for (int ks = 0; ks < 64; ks += 32) {
    v16bf a0 = load_a_frag(&Qs[(wm * 32) * 64 + ks], 64, 1);
    v16bf a1 = load_a_frag(&Qs[(wm * 32 + 16) * 64 + ks], 64, 1);
    // B[k][n=e] = KVt[(e0+n)*64 + k]  (contiguous in k)
    v16bf b0 = load_b_frag(&KVt[(wn * 32) * 64 + ks], 1, 64);
    v16bf b1 = load_b_frag(&KVt[(wn * 32 + 16) * 64 + ks], 1, 64);
    acc[0][0] = wmma_bf16(a0, b0, acc[0][0]);
    acc[0][1] = wmma_bf16(a0, b1, acc[0][1]);
    acc[1][0] = wmma_bf16(a1, b0, acc[1][0]);
    acc[1][1] = wmma_bf16(a1, b1, acc[1][1]);
  }

  const int lane = tid & 31, nloc = lane & 15, half = lane >> 4;
  const int bi = bh >> 3, hh = bh & 7;
#pragma unroll
  for (int i = 0; i < 2; ++i)
#pragma unroll
    for (int j = 0; j < 2; ++j)
#pragma unroll
      for (int r = 0; r < 8; ++r) {
        const int m = wm * 32 + i * 16 + r + 8 * half;
        const int e = wn * 32 + j * 16 + nloc;
        const float v = acc[i][j][r] * zl[m];
        attn[(size_t)(bi * N_ + n0 + m) * INNER_ + hh * HEAD_DIM + e] = (bf16_t)v;
      }
}

// ---------------------------------------------------------------------------
// K4: y = attn @ W_out^T + b_out (fp32 out). 128x128 tiles, async staging.
// ---------------------------------------------------------------------------
__global__ __launch_bounds__(256) void out_gemm_kernel(
    const bf16_t* __restrict__ attn, const bf16_t* __restrict__ wout,
    const float* __restrict__ bout, float* __restrict__ out) {
  __shared__ bf16_t As[128 * 32];
  __shared__ bf16_t Bs[128 * 32];

  const int tid = threadIdx.x;
  const int m0 = blockIdx.x * 128;
  const int j0 = blockIdx.y * 128;
  const int wid = tid >> 5;
  const int wm = wid & 3, wn = wid >> 2;

  v8f acc[2][4] = {};

  for (int k0 = 0; k0 < INNER_; k0 += 32) {
#pragma unroll
    for (int t = 0; t < 2; ++t) {
      const int idx = tid + t * 256;
      const int row = idx >> 2, c = (idx & 3) * 8;
      async_copy_b128(attn + (size_t)(m0 + row) * INNER_ + k0 + c,
                      &As[row * 32 + c]);
      async_copy_b128(wout + (size_t)(j0 + row) * INNER_ + k0 + c,
                      &Bs[row * 32 + c]);
    }
    async_wait0();
    __syncthreads();

    v16bf a0 = load_a_frag(&As[(wm * 32) * 32], 32, 1);
    v16bf a1 = load_a_frag(&As[(wm * 32 + 16) * 32], 32, 1);
#pragma unroll
    for (int j = 0; j < 4; ++j) {
      v16bf b = load_b_frag(&Bs[(wn * 64 + j * 16) * 32], 1, 32);
      acc[0][j] = wmma_bf16(a0, b, acc[0][j]);
      acc[1][j] = wmma_bf16(a1, b, acc[1][j]);
    }
    __syncthreads();
  }

  const int lane = tid & 31, nloc = lane & 15, half = lane >> 4;
#pragma unroll
  for (int i = 0; i < 2; ++i)
#pragma unroll
    for (int j = 0; j < 4; ++j)
#pragma unroll
      for (int r = 0; r < 8; ++r) {
        const int M = m0 + wm * 32 + i * 16 + r + 8 * half;
        const int J = j0 + wn * 64 + j * 16 + nloc;
        out[(size_t)M * DIM_ + J] = acc[i][j][r] + bout[J];
      }
}

// ---------------------------------------------------------------------------
extern "C" void kernel_launch(void* const* d_in, const int* in_sizes, int n_in,
                              void* d_out, int out_size, void* d_ws,
                              size_t ws_size, hipStream_t stream) {
  const float* x     = (const float*)d_in[0];
  const float* W_qkv = (const float*)d_in[1];
  const float* W_out = (const float*)d_in[2];
  const float* b_out = (const float*)d_in[3];

  char* ws = (char*)d_ws;
  bf16_t* qh      = (bf16_t*)(ws + OFF_QH);
  bf16_t* kh      = (bf16_t*)(ws + OFF_KH);
  bf16_t* vh      = (bf16_t*)(ws + OFF_VH);
  bf16_t* attn    = (bf16_t*)(ws + OFF_ATTN);
  bf16_t* wqkv_bf = (bf16_t*)(ws + OFF_WQKV);
  bf16_t* wout_bf = (bf16_t*)(ws + OFF_WOUT);
  float*  kvbuf   = (float*)(ws + OFF_KV);
  float*  ksumbuf = (float*)(ws + OFF_KSUM);

  const int nwq = 3 * INNER_ * DIM_;
  const int nwo = DIM_ * INNER_;
  cvt_f32_bf16_kernel<<<(nwq + 255) / 256, 256, 0, stream>>>(W_qkv, wqkv_bf, nwq);
  cvt_f32_bf16_kernel<<<(nwo + 255) / 256, 256, 0, stream>>>(W_out, wout_bf, nwo);

  // K1: [65536 x 1536], 128x128 tiles
  qkv_gemm_kernel<<<dim3(ROWS_ / 128, (3 * INNER_) / 128), 256, 0, stream>>>(
      x, wqkv_bf, qh, kh, vh);

  // K2: one block per (b,h)
  kv_kernel<<<B_ * NUM_HEADS, 256, 0, stream>>>(kh, vh, kvbuf, ksumbuf);

  // K3: (n/128) x (b*h)
  attn_out_kernel<<<dim3(N_ / 128, B_ * NUM_HEADS), 256, 0, stream>>>(
      qh, kvbuf, ksumbuf, attn);

  // K4: [65536 x 512], 128x128 tiles
  out_gemm_kernel<<<dim3(ROWS_ / 128, DIM_ / 128), 256, 0, stream>>>(
      attn, wout_bf, b_out, (float*)d_out);
}
